// GCN1_39539468927054
// MI455X (gfx1250) — matrix operand order repaired and checked by reading, not compiled, not run on hardware
//
#include <hip/hip_runtime.h>
#include <hip/hip_bf16.h>

typedef __attribute__((ext_vector_type(16))) __bf16 v16bf;
typedef __attribute__((ext_vector_type(8)))  float  v8f;
typedef __attribute__((ext_vector_type(4)))  float  v4f;

#define N_DIM 8192
#define DDIM 512

// ---------------------------------------------------------------------------
// Workspace layout (bytes):
//   [0, 16K)          : hist[4096] u32
//   [16K, 16K+256)    : state[] u32  (0: prefix/key, 1: residual rank, 2: median bits)
//   [16K+256, +32K)   : rowmax[8192] f32
//   [.. +32K)         : rowsum[8192] f32
//   [..]              : fcXT bf16 [512][8192]   (8 MB)
// fc2X (fp32, 16 MB) lives in d_out and is overwritten in-place by the epilogue.
// ---------------------------------------------------------------------------

__global__ void zero_hist_kernel(unsigned* hist) {
    int i = blockIdx.x * blockDim.x + threadIdx.x;
    if (i < 4096) hist[i] = 0u;
}

// Histogram of `bits`-wide digit at `shift` of the float bit pattern, over the
// strictly-upper-triangular entries of A, restricted (when use_prefix) to keys
// whose bits above (shift+bits) equal state[0].  A is streamed non-temporally.
__global__ __launch_bounds__(256) void hist_pass_kernel(const float* __restrict__ A,
                                                        unsigned* __restrict__ hist,
                                                        const unsigned* __restrict__ state,
                                                        int shift, int bits, int use_prefix) {
    __shared__ unsigned lh[4096];
    const int nb = 1 << bits;
    for (int i = threadIdx.x; i < nb; i += blockDim.x) lh[i] = 0u;
    __syncthreads();

    const unsigned prefix = use_prefix ? state[0] : 0u;
    const int top = shift + bits;
    const size_t total = (size_t)N_DIM * N_DIM;
    const size_t stride = (size_t)gridDim.x * blockDim.x;
    for (size_t idx = (size_t)blockIdx.x * blockDim.x + threadIdx.x; idx < total; idx += stride) {
        unsigned i = (unsigned)(idx >> 13);
        unsigned j = (unsigned)(idx & (N_DIM - 1));
        if (j <= i) continue;                       // strict upper triangle only
        float a = __builtin_nontemporal_load(A + idx);
        unsigned key = __float_as_uint(a);          // positive floats: order-preserving bits
        unsigned hi = (top >= 32) ? 0u : (key >> top);
        if (!use_prefix || hi == prefix)
            atomicAdd(&lh[(key >> shift) & (nb - 1)], 1u);
    }
    __syncthreads();
    for (int i = threadIdx.x; i < nb; i += blockDim.x)
        if (lh[i]) atomicAdd(&hist[i], lh[i]);
}

// Single-thread digit selection; re-zeroes hist so the launch sequence is
// idempotent across graph replays.
__global__ void scan_select_kernel(unsigned* hist, unsigned* state,
                                   int bits, int is_first, int is_final) {
    if (blockIdx.x != 0 || threadIdx.x != 0) return;
    unsigned long long target;
    unsigned prefix;
    if (is_first) {
        const unsigned long long m = (unsigned long long)N_DIM * (N_DIM - 1) / 2;
        target = (m - 1) / 2;                        // lower median rank
        prefix = 0u;
    } else {
        target = state[1];
        prefix = state[0];
    }
    const int nb = 1 << bits;
    unsigned long long cum = 0;
    int b = nb - 1;
    for (int i = 0; i < nb; ++i) {
        unsigned c = hist[i];
        if (cum + c > target) { b = i; target -= cum; break; }
        cum += c;
    }
    state[0] = (prefix << bits) | (unsigned)b;
    state[1] = (unsigned)target;
    if (is_final) state[2] = state[0];               // full 32-bit pattern of the median element
    for (int i = 0; i < nb; ++i) hist[i] = 0u;       // ready for next pass / next replay
}

// Per-row max and sum(exp(thr(a)-max)) for the softmax, one WG per row,
// row cached in LDS (32 KB of the 320 KB WGP budget).
__global__ __launch_bounds__(256) void row_stats_kernel(const float* __restrict__ A,
                                                        const unsigned* __restrict__ state,
                                                        float* __restrict__ rowmax,
                                                        float* __restrict__ rowsum) {
    __shared__ float buf[N_DIM];
    __shared__ float red[256];
    const float med = __uint_as_float(state[2]);
    const int row = blockIdx.x;
    const float* ar = A + (size_t)row * N_DIM;

    float mx = 0.0f;                                 // thresholded values are >= 0
    for (int j = threadIdx.x; j < N_DIM; j += 256) {
        float a = __builtin_nontemporal_load(ar + j);
        a = (a < med) ? 0.0f : a;
        buf[j] = a;
        mx = fmaxf(mx, a);
    }
    red[threadIdx.x] = mx; __syncthreads();
    for (int s = 128; s > 0; s >>= 1) {
        if (threadIdx.x < s) red[threadIdx.x] = fmaxf(red[threadIdx.x], red[threadIdx.x + s]);
        __syncthreads();
    }
    mx = red[0]; __syncthreads();

    float sum = 0.0f;
    for (int j = threadIdx.x; j < N_DIM; j += 256) sum += __expf(buf[j] - mx);
    red[threadIdx.x] = sum; __syncthreads();
    for (int s = 128; s > 0; s >>= 1) {
        if (threadIdx.x < s) red[threadIdx.x] += red[threadIdx.x + s];
        __syncthreads();
    }
    if (threadIdx.x == 0) { rowmax[row] = mx; rowsum[row] = red[0]; }
}

// ---------------------------------------------------------------------------
// WMMA tiling: 32 M-rows per WG, 8 waves, each wave = 32x64 outputs
// = 2 (M) x 4 (N) tiles of 16x16.
// A-fragment staging: lane L holds row M=L&15; elem e -> K = (L>>4)*16 + e
// (contiguous).  Inverse: h=k>>4, e=k&15.
// B-fragment: lane L holds col N=L&15; elem e -> K = (L>>4)*16 + e (contiguous).
// ---------------------------------------------------------------------------

__device__ __forceinline__ void stage_a_elem(__bf16* ldsA /*[2][512]*/, int row, int k, float v) {
    int h = k >> 4;
    int e = k & 15;
    int lane = h * 16 + (row & 15);
    ldsA[(row >> 4) * 512 + lane * 16 + e] = (__bf16)v;
}

// fcX / fc2X: out[m][o] = sum_k X[m][k]*W[o][k] + b[o].
// mode 0: write bf16 transposed outT[o][m]; mode 1: write fp32 out_f32[m][o].
__global__ __launch_bounds__(256) void fc_gemm_kernel(const float* __restrict__ X,
                                                      const float* __restrict__ W,
                                                      const float* __restrict__ bias,
                                                      __bf16* __restrict__ outT,
                                                      float* __restrict__ out_f32,
                                                      int mode) {
    __shared__ __bf16 ldsA[2 * 512];
    const int i0 = blockIdx.x * 32;
    const int tid = threadIdx.x;
    const int wave = tid >> 5;
    const int lane = tid & 31;
    const int n_base = wave * 64;

    v8f acc[2][4] = {};

    const int arow = tid >> 3;          // 0..31
    const int acol0 = (tid & 7) * 4;    // 0..28 step 4

    for (int k0 = 0; k0 < DDIM; k0 += 32) {
        v4f av = *(const v4f*)(X + (size_t)(i0 + arow) * DDIM + k0 + acol0);
        __syncthreads();
#pragma unroll
        for (int q = 0; q < 4; ++q) stage_a_elem(ldsA, arow, acol0 + q, av[q]);
        __syncthreads();

        v16bf afrag0 = *(const v16bf*)&ldsA[0 * 512 + lane * 16];
        v16bf afrag1 = *(const v16bf*)&ldsA[1 * 512 + lane * 16];

        const int h = lane >> 4;
#pragma unroll
        for (int nt = 0; nt < 4; ++nt) {
            const int o = n_base + nt * 16 + (lane & 15);
            const v4f* wp = (const v4f*)(W + (size_t)o * DDIM + k0 + h * 16);
            v16bf bfrag;
#pragma unroll
            for (int q = 0; q < 4; ++q) {
                v4f w = wp[q];
                bfrag[4 * q + 0] = (__bf16)w[0];
                bfrag[4 * q + 1] = (__bf16)w[1];
                bfrag[4 * q + 2] = (__bf16)w[2];
                bfrag[4 * q + 3] = (__bf16)w[3];
            }
            acc[0][nt] = __builtin_amdgcn_wmma_f32_16x16x32_bf16(false, afrag0, false, bfrag,
                                                                 (short)0, acc[0][nt], false, false);
            acc[1][nt] = __builtin_amdgcn_wmma_f32_16x16x32_bf16(false, afrag1, false, bfrag,
                                                                 (short)0, acc[1][nt], false, false);
        }
    }

    // C/D layout: VGPR v, lane L -> M = v + (L>>4)*8, N = L&15
#pragma unroll
    for (int mt = 0; mt < 2; ++mt)
#pragma unroll
        for (int nt = 0; nt < 4; ++nt)
#pragma unroll
            for (int v = 0; v < 8; ++v) {
                int m = i0 + mt * 16 + (lane >> 4) * 8 + v;
                int o = n_base + nt * 16 + (lane & 15);
                float val = acc[mt][nt][v] + bias[o];
                if (mode == 0) outT[(size_t)o * N_DIM + m] = (__bf16)val;
                else           out_f32[(size_t)m * DDIM + o] = val;
            }
}

// Fused: out[m][n] = leaky( 0.1*fc2X[m][n] + (0.9/rowsum[m]) *
//                            sum_j exp(thr(A[m][j]) - rowmax[m]) * fcXT[n][j] )
// fc2X lives in `out` and is overwritten in place.  K-step 64 (two 32-K
// halves staged per barrier pair -> 16 WMMAs per wave between barriers).
__global__ __launch_bounds__(256) void norm_gemm_kernel(const float* __restrict__ A,
                                                        const __bf16* __restrict__ fcXT,
                                                        const float* __restrict__ rowmax,
                                                        const float* __restrict__ rowsum,
                                                        const unsigned* __restrict__ state,
                                                        float* __restrict__ out) {
    __shared__ __bf16 ldsA[2 * 1024];                // [khalf][subtile][512]
    const float med = __uint_as_float(state[2]);
    const int i0 = blockIdx.x * 32;
    const int tid = threadIdx.x;
    const int wave = tid >> 5;
    const int lane = tid & 31;
    const int n_base = wave * 64;

    v8f acc[2][4] = {};

    const int arow = tid >> 3;
    const int acol0 = (tid & 7) * 4;
    const float rmax = rowmax[i0 + arow];
    const float* arp = A + (size_t)(i0 + arow) * N_DIM + acol0;

    for (int k0 = 0; k0 < N_DIM; k0 += 64) {
        v4f av0 = __builtin_nontemporal_load((const v4f*)(arp + k0));        // NT: spare L2 for fcXT
        v4f av1 = __builtin_nontemporal_load((const v4f*)(arp + k0 + 32));
        __builtin_prefetch(arp + k0 + 128, 0, 0);    // global_prefetch_b8 down the A stream
        __syncthreads();
#pragma unroll
        for (int q = 0; q < 4; ++q) {
            float a0 = av0[q]; a0 = (a0 < med) ? 0.0f : a0;
            float a1 = av1[q]; a1 = (a1 < med) ? 0.0f : a1;
            stage_a_elem(ldsA,        arow, acol0 + q, __expf(a0 - rmax));   // K half 0
            stage_a_elem(ldsA + 1024, arow, acol0 + q, __expf(a1 - rmax));   // K half 1
        }
        __syncthreads();

        v16bf af00 = *(const v16bf*)&ldsA[0 * 1024 + 0 * 512 + lane * 16];   // kh0, rows 0-15
        v16bf af01 = *(const v16bf*)&ldsA[0 * 1024 + 1 * 512 + lane * 16];   // kh0, rows 16-31
        v16bf af10 = *(const v16bf*)&ldsA[1 * 1024 + 0 * 512 + lane * 16];   // kh1, rows 0-15
        v16bf af11 = *(const v16bf*)&ldsA[1 * 1024 + 1 * 512 + lane * 16];   // kh1, rows 16-31

        const int h = lane >> 4;
#pragma unroll
        for (int nt = 0; nt < 4; ++nt) {
            const int n = n_base + nt * 16 + (lane & 15);
            const __bf16* bp = fcXT + (size_t)n * N_DIM + k0 + h * 16;
            v16bf b0 = *(const v16bf*)bp;
            v16bf b1 = *(const v16bf*)(bp + 32);
            acc[0][nt] = __builtin_amdgcn_wmma_f32_16x16x32_bf16(false, af00, false, b0,
                                                                 (short)0, acc[0][nt], false, false);
            acc[1][nt] = __builtin_amdgcn_wmma_f32_16x16x32_bf16(false, af01, false, b0,
                                                                 (short)0, acc[1][nt], false, false);
            acc[0][nt] = __builtin_amdgcn_wmma_f32_16x16x32_bf16(false, af10, false, b1,
                                                                 (short)0, acc[0][nt], false, false);
            acc[1][nt] = __builtin_amdgcn_wmma_f32_16x16x32_bf16(false, af11, false, b1,
                                                                 (short)0, acc[1][nt], false, false);
        }
    }

#pragma unroll
    for (int mt = 0; mt < 2; ++mt)
#pragma unroll
        for (int nt = 0; nt < 4; ++nt) {
            int m0 = i0 + mt * 16 + (lane >> 4) * 8;
            int n = n_base + nt * 16 + (lane & 15);
#pragma unroll
            for (int v = 0; v < 8; ++v) {
                int m = m0 + v;
                float scale = 0.9f / rowsum[m];
                size_t idx = (size_t)m * DDIM + n;
                float val = 0.1f * out[idx] + scale * acc[mt][nt][v];
                out[idx] = (val >= 0.0f) ? val : 0.01f * val;
            }
        }
}

extern "C" void kernel_launch(void* const* d_in, const int* in_sizes, int n_in,
                              void* d_out, int out_size, void* d_ws, size_t ws_size,
                              hipStream_t stream) {
    const float* A  = (const float*)d_in[0];
    const float* X  = (const float*)d_in[1];
    const float* W1 = (const float*)d_in[2];
    const float* b1 = (const float*)d_in[3];
    const float* W2 = (const float*)d_in[4];
    const float* b2 = (const float*)d_in[5];
    float* out = (float*)d_out;

    char* ws = (char*)d_ws;
    unsigned* hist   = (unsigned*)ws;                              // 16 KB
    unsigned* state  = (unsigned*)(ws + 16 * 1024);                // few words
    float*    rowmax = (float*)(ws + 16 * 1024 + 256);             // 32 KB
    float*    rowsum = rowmax + N_DIM;                             // 32 KB
    __bf16*   fcXT   = (__bf16*)(ws + 16 * 1024 + 256 + 2 * N_DIM * 4);  // 8 MB

    // 1) exact lower-median of strict-upper-tri(A) via 3-pass radix select (12+12+8 bits)
    zero_hist_kernel<<<16, 256, 0, stream>>>(hist);
    hist_pass_kernel<<<2048, 256, 0, stream>>>(A, hist, state, 20, 12, 0);
    scan_select_kernel<<<1, 32, 0, stream>>>(hist, state, 12, 1, 0);
    hist_pass_kernel<<<2048, 256, 0, stream>>>(A, hist, state, 8, 12, 1);
    scan_select_kernel<<<1, 32, 0, stream>>>(hist, state, 12, 0, 0);
    hist_pass_kernel<<<2048, 256, 0, stream>>>(A, hist, state, 0, 8, 1);
    scan_select_kernel<<<1, 32, 0, stream>>>(hist, state, 8, 0, 1);

    // 2) per-row softmax stats of thresholded A
    row_stats_kernel<<<N_DIM, 256, 0, stream>>>(A, state, rowmax, rowsum);

    // 3) fcX -> bf16 transposed (GEMM B operand); fc2X -> fp32 into d_out
    fc_gemm_kernel<<<N_DIM / 32, 256, 0, stream>>>(X, W1, b1, fcXT, nullptr, 0);
    fc_gemm_kernel<<<N_DIM / 32, 256, 0, stream>>>(X, W2, b2, nullptr, out, 1);

    // 4) fused softmax-normalized GEMM + epilogue (overwrites fc2X in d_out)
    norm_gemm_kernel<<<N_DIM / 32, 256, 0, stream>>>(A, fcXT, rowmax, rowsum, state, out);
}
